// BitConv1d_14912126451940
// MI455X (gfx1250) — matrix operand order repaired and checked by reading, not compiled, not run on hardware
//
#include <hip/hip_runtime.h>
#include <hip/hip_bf16.h>

// ---------------------------------------------------------------------------
// BitConv1d on gfx1250:  out = conv1d(x, sign(w), pad=1) * mean(|w|) * scale
// (per-sample x_scale cancels algebraically: conv(x/s, w)*s == conv(x, w))
//
// GEMM per sample n:  Out[512,4096] = Wbin[512,1536] @ Xim2col[1536,4096]
// K-major flat k-index j = tap*512 + ci  ->  each 64-chunk lives in one tap.
// Weights binarized ONCE into ws as bf16 [co][j]; per-chunk A tiles are DMA'd
// into LDS by the Tensor Data Mover; x tiles converted f32->bf16 in VALU.
// Core: v_wmma_f32_16x16x32_bf16, f32 accumulation.
// ---------------------------------------------------------------------------

typedef __attribute__((ext_vector_type(16))) __bf16 v16bf;
typedef __attribute__((ext_vector_type(8)))  __bf16 v8bf;
typedef __attribute__((ext_vector_type(8)))  float  v8f;
typedef __attribute__((ext_vector_type(4)))  unsigned int u32x4;
typedef __attribute__((ext_vector_type(8)))  int          i32x8;
typedef __attribute__((ext_vector_type(4)))  int          i32x4;

#define CIN   512
#define COUT  512
#define LEN   4096
#define KTAPS 3
#define KTOT  (CIN * KTAPS)   // 1536
#define BM    128             // C_out tile
#define BN    128             // L tile
#define KC    64              // K chunk (divides 512 -> never straddles a tap)

#if __has_builtin(__builtin_amdgcn_tensor_load_to_lds)
#define USE_TDM 1
#else
#define USE_TDM 0
#endif

union AFrag { v16bf v; v8bf h[2]; };

// ---- Kernel 1: w_scale = mean(|w|), single-workgroup tree reduction --------
__global__ void __launch_bounds__(1024)
wscale_kernel(const float* __restrict__ w, float* __restrict__ ws) {
    __shared__ float red[1024];
    const int NW = COUT * CIN * KTAPS;
    float s = 0.f;
    for (int i = threadIdx.x; i < NW; i += 1024) s += fabsf(w[i]);
    red[threadIdx.x] = s;
    __syncthreads();
    for (int off = 512; off > 0; off >>= 1) {
        if ((int)threadIdx.x < off) red[threadIdx.x] += red[threadIdx.x + off];
        __syncthreads();
    }
    if (threadIdx.x == 0) ws[0] = red[0] / (float)NW;
}

// ---- Kernel 2: binarize weights once: wbin[co][j] = sign(w[co][ci][tap]) ---
// j = tap*512 + ci (K-major), bf16 +1/-1/0 -> every A tile row is contiguous.
__global__ void __launch_bounds__(256)
wbin_kernel(const float* __restrict__ w, unsigned short* __restrict__ wbin) {
    const int idx = blockIdx.x * 256 + threadIdx.x;   // [0, 512*1536)
    const int co  = idx / KTOT;
    const int j   = idx - co * KTOT;
    const int tap = j >> 9, ci = j & (CIN - 1);
    const float wv = w[(size_t)co * KTOT + ci * KTAPS + tap];
    wbin[idx] = (wv > 0.f) ? (unsigned short)0x3F80u
                           : (wv < 0.f ? (unsigned short)0xBF80u : (unsigned short)0u);
}

// ---- Kernel 3: tiled bf16 WMMA GEMM, TDM-staged A, VALU-converted B --------
__global__ void __launch_bounds__(256)
bitconv_wmma_kernel(const float* __restrict__ x,
                    const unsigned short* __restrict__ wbin,
                    const float* __restrict__ scale,
                    const float* __restrict__ wsum,
                    float* __restrict__ out) {
    __shared__ unsigned short sA[BM * KC];  // sign(w) bf16, [m][k], 16 KB
    __shared__ __bf16         sB[BN * KC];  // x bf16, transposed [l][k], 16 KB

    const int n      = blockIdx.z;
    const int coBase = blockIdx.y * BM;
    const int lBase  = blockIdx.x * BN;
    const int tid    = threadIdx.x;
    const int lane   = tid & 31;
    const int wid    = tid >> 5;
    const int Moff   = (wid & 3) * 32;    // 4 waves along M
    const int Noff   = (wid >> 2) * 64;   // 2 waves along N
    const int half   = lane >> 4;
    const int lr     = lane & 15;

    const float* xN = x + (size_t)n * CIN * LEN;
    const size_t wRowBase = (size_t)coBase * KTOT;

#if USE_TDM
    // Generic->LDS pointers carry the LDS byte offset in the low 32 bits.
    const unsigned ldsA = (unsigned)(uintptr_t)(void*)sA;
#endif

    v8f acc[2][4] = {};

    for (int kk = 0; kk < KTOT; kk += KC) {
        const int tap    = kk >> 9;         // kk / 512
        const int ciBase = kk & (CIN - 1);  // kk % 512

        // --- stage A: 128 rows x 64 bf16, row stride KTOT, via TDM ----------
#if USE_TDM
        if (wid == 0) {
            const unsigned long long gaddr =
                (unsigned long long)(uintptr_t)(wbin + wRowBase + kk);
            u32x4 g0;
            g0[0] = 1u;                              // count=1, user mode
            g0[1] = ldsA;                            // lds_addr (bytes)
            g0[2] = (unsigned)(gaddr & 0xFFFFFFFFu); // global_addr[31:0]
            g0[3] = (unsigned)((gaddr >> 32) & 0x1FFFFFFu) | (2u << 30); // type=2
            i32x8 g1;
            g1[0] = (int)(1u << 16);                 // data_size=2B; no mask
            g1[1] = (int)((unsigned)KC << 16);       // tensor_dim0[15:0]=64
            g1[2] = (int)((unsigned)BM << 16);       // tensor_dim1[15:0]=128
            g1[3] = (int)((unsigned)KC << 16);       // tile_dim0=64
            g1[4] = (int)(unsigned)BM;               // tile_dim1=128, tile_dim2=0
            g1[5] = (int)(unsigned)KTOT;             // tensor_dim0_stride=1536
            g1[6] = 0;                               // stride0 hi / stride1 lo
            g1[7] = 0;                               // stride1 hi
            const i32x4 gz4 = {};
            const i32x8 gz8 = {};
            __builtin_amdgcn_tensor_load_to_lds(g0, g1, gz4, gz4, gz8, 0);
            __builtin_amdgcn_s_wait_tensorcnt(0);
        }
#else
        // Fallback: vector-copy A tile (still no binarization in the loop).
        #pragma unroll
        for (int i = 0; i < (BM * KC * 2) / (256 * 16); ++i) {   // 4 x 16B/thread
            const int idx = tid + i * 256;                       // 1024 chunks
            const int row = idx >> 3, col = (idx & 7) * 8;
            ((uint4*)sA)[idx] =
                *(const uint4*)(wbin + (size_t)(coBase + row) * KTOT + kk + col);
        }
#endif

        // --- stage B: x tile transposed [l][ci], coalesced on l, hw cvt -----
        #pragma unroll
        for (int i = 0; i < (BN * KC) / 256; ++i) {
            const int idx  = tid + i * 256;
            const int ll   = idx & (BN - 1);
            const int t    = idx >> 7;               // 0..63
            const int lsrc = lBase + ll + tap - 1;   // pad=1 shift
            const float xv = (lsrc >= 0 && lsrc < LEN)
                                 ? xN[(size_t)(ciBase + t) * LEN + lsrc]
                                 : 0.f;
            sB[ll * KC + t] = (__bf16)xv;
        }
        __syncthreads();

        // --- 2 WMMA K-steps of 32 over this chunk ---------------------------
        #pragma unroll
        for (int t = 0; t < KC; t += 32) {
            // A (16x32 bf16): lanes 0-15 hold K[t..t+15];
            //                 lanes 16-31 hold K[t+16..t+31].
            AFrag a[2];
            #pragma unroll
            for (int mi = 0; mi < 2; ++mi) {
                const unsigned short* row = &sA[(Moff + mi * 16 + lr) * KC];
                a[mi].h[0] = *(const v8bf*)(row + t + half * 16);
                a[mi].h[1] = *(const v8bf*)(row + t + half * 16 + 8);
            }
            // B (32x16 bf16): lane half selects K[t..t+15] / K[t+16..t+31]
            // of column lr -> contiguous in transposed sB.
            v16bf b[4];
            #pragma unroll
            for (int ni = 0; ni < 4; ++ni)
                b[ni] = *(const v16bf*)&sB[(Noff + ni * 16 + lr) * KC + t + half * 16];

            #pragma unroll
            for (int mi = 0; mi < 2; ++mi)
                #pragma unroll
                for (int ni = 0; ni < 4; ++ni)
                    acc[mi][ni] = __builtin_amdgcn_wmma_f32_16x16x32_bf16(
                        false, a[mi].v, false, b[ni],
                        (short)0, acc[mi][ni], false, false);
        }
        __syncthreads();
    }

    // --- epilogue: D layout -> VGPR r: M = r + 8*half, N = lr ---------------
    const float g = wsum[0] * scale[0];   // x_scale cancelled analytically
    #pragma unroll
    for (int mi = 0; mi < 2; ++mi) {
        #pragma unroll
        for (int ni = 0; ni < 4; ++ni) {
            const int l = lBase + Noff + ni * 16 + lr;
            #pragma unroll
            for (int r = 0; r < 8; ++r) {
                const int co = coBase + Moff + mi * 16 + r + half * 8;
                out[((size_t)n * COUT + co) * (size_t)LEN + l] = acc[mi][ni][r] * g;
            }
        }
    }
}

// ---------------------------------------------------------------------------
extern "C" void kernel_launch(void* const* d_in, const int* in_sizes, int n_in,
                              void* d_out, int out_size, void* d_ws, size_t ws_size,
                              hipStream_t stream) {
    (void)n_in; (void)out_size; (void)ws_size;
    const float* x     = (const float*)d_in[0];   // [N, 512, 4096] f32
    const float* w     = (const float*)d_in[1];   // [512, 512, 3]  f32
    const float* scale = (const float*)d_in[2];   // [1] f32
    float*       out   = (float*)d_out;           // [N, 512, 4096] f32

    float*          wsum = (float*)d_ws;                            // 4 B
    unsigned short* wbin = (unsigned short*)((char*)d_ws + 256);    // 1.5 MB

    const int NBATCH = in_sizes[0] / (CIN * LEN); // 16

    wscale_kernel<<<1, 1024, 0, stream>>>(w, wsum);
    wbin_kernel<<<(COUT * KTOT) / 256, 256, 0, stream>>>(w, wbin);

    dim3 grid(LEN / BN, COUT / BM, NBATCH);       // (32, 4, 16)
    bitconv_wmma_kernel<<<grid, 256, 0, stream>>>(x, wbin, scale, wsum, out);
}
